// WindowCrossAttention_39204461478453
// MI455X (gfx1250) — compile-verified, matrix operand order checked by reading, not run on hardware
//
#include <hip/hip_runtime.h>
#include <hip/hip_bf16.h>
#include <stdint.h>

typedef __bf16 bf16;
typedef __attribute__((ext_vector_type(16))) __bf16 v16bf;
typedef __attribute__((ext_vector_type(8)))  __bf16 v8bf;
typedef __attribute__((ext_vector_type(8)))  float  v8f;
typedef __attribute__((ext_vector_type(4)))  float  v4f;

// Global-address-space pointer types: force global_load/global_store (not flat_*,
// which would also tie up DScnt / the LDS path).
typedef const __attribute__((address_space(1))) v8bf* gcv8bf;
typedef const __attribute__((address_space(1))) v4f*  gcv4f;
typedef __attribute__((address_space(1))) bf16*       gbf16p;
typedef __attribute__((address_space(1))) float*      gf32p;

// LDS fragment: two contiguous 16B chunks (compiler sees AS3 -> ds_load_b128).
__device__ __forceinline__ v16bf lds_frag(const bf16* p0) {
    union { v16bf v; v8bf h[2]; } u;
    u.h[0] = *(const v8bf*)p0;
    u.h[1] = *(const v8bf*)(p0 + 16);
    return u.v;
}

// Global fragment: two contiguous 16B chunks via addrspace(1) -> global_load_b128.
__device__ __forceinline__ v16bf gbl_frag(const bf16* p0) {
    union { v16bf v; v8bf h[2]; } u;
    u.h[0] = *(gcv8bf)(uintptr_t)(p0);
    u.h[1] = *(gcv8bf)(uintptr_t)(p0 + 16);
    return u.v;
}

// A-fragment from an fp32 row (global): k-chunks [0..7] and [16..23], cvt to bf16.
__device__ __forceinline__ v16bf gbl_fragA_f32(const float* pa) {
    union { v16bf v; __bf16 e[16]; } a;
    v4f f0 = *(gcv4f)(uintptr_t)(pa);
    v4f f1 = *(gcv4f)(uintptr_t)(pa + 4);
    v4f f2 = *(gcv4f)(uintptr_t)(pa + 16);
    v4f f3 = *(gcv4f)(uintptr_t)(pa + 20);
    #pragma unroll
    for (int i = 0; i < 4; i++) {
        a.e[i]      = (__bf16)f0[i];
        a.e[4 + i]  = (__bf16)f1[i];
        a.e[8 + i]  = (__bf16)f2[i];
        a.e[12 + i] = (__bf16)f3[i];
    }
    return a.v;
}

// ---------------- Kernel: fp32 -> bf16 weight conversion ----------------
__global__ void cvt_f32_bf16(const float* __restrict__ src, bf16* __restrict__ dst, int n) {
    int i = blockIdx.x * 256 + threadIdx.x;
    if (i < n) dst[i] = (__bf16)src[i];
}

// ---------------- Kernel: pack rel-pos bias + mask, padded (64,16,64,128) ----------------
__global__ void bias_pack(const float* __restrict__ bias_table, const int* __restrict__ rel_index,
                          const float* __restrict__ mask, float* __restrict__ biasfull) {
    int idx = blockIdx.x * 256 + threadIdx.x;
    const int total = 64 * 16 * 64 * 128;
    if (idx >= total) return;
    int n = idx & 127;
    int m = (idx >> 7) & 63;
    int h = (idx >> 13) & 15;
    int w = idx >> 17;
    float val = -1.0e30f;                      // padded rows/cols -> exp() == 0
    if (m < 49 && n < 98) {
        int nk = (n < 49) ? n : n - 49;        // rpb / mask concatenated along keys
        int ri = rel_index[m * 49 + nk];
        val = bias_table[ri * 16 + h] + mask[(w * 49 + m) * 49 + nk];
    }
    biasfull[idx] = val;
}

// ---------------- Kernel: GEMM  C_bf16 = (A_f32 @ W^T + bias) * scale ----------------
// Double-buffered K loop: next-step A/B fragments load while current WMMAs issue.
__global__ __launch_bounds__(128)
void gemm_a32_cbf16(const float* __restrict__ A0, const float* __restrict__ A1,
                    int rows_per_b, int split,
                    const bf16* __restrict__ W, const float* __restrict__ bias,
                    bf16* __restrict__ C, int M_total, int Nout, float scale) {
    const int lane  = threadIdx.x & 31;
    const int wave  = threadIdx.x >> 5;
    const int mtile = blockIdx.x * 4 + wave;
    const int n0    = blockIdx.y * 64;
    if (mtile * 16 >= M_total) return;
    const int mloc  = lane & 15;
    const int khalf = lane >> 4;

    int m  = mtile * 16 + mloc;
    int mc = min(m, M_total - 1);
    int b  = mc / rows_per_b, t = mc % rows_per_b;
    const float* arow = (t < split)
        ? (A0 + ((size_t)b * split + t) * 512)
        : (A1 + ((size_t)b * (rows_per_b - split) + (t - split)) * 512);

    const float* aptr = arow + khalf * 8;
    const bf16*  wptr[4];
    #pragma unroll
    for (int nt = 0; nt < 4; nt++)
        wptr[nt] = W + (size_t)(n0 + nt * 16 + mloc) * 512 + khalf * 8;

    // prologue
    v16bf a_cur = gbl_fragA_f32(aptr);
    v16bf b_cur[4];
    #pragma unroll
    for (int nt = 0; nt < 4; nt++) b_cur[nt] = gbl_frag(wptr[nt]);

    v8f acc[4] = {};
    #pragma unroll 3
    for (int k0 = 0; k0 < 480; k0 += 32) {
        aptr += 32;
        __builtin_prefetch(aptr + 96, 0, 1);
        v16bf a_nxt = gbl_fragA_f32(aptr);
        v16bf b_nxt[4];
        #pragma unroll
        for (int nt = 0; nt < 4; nt++) {
            wptr[nt] += 32;
            b_nxt[nt] = gbl_frag(wptr[nt]);
        }
        #pragma unroll
        for (int nt = 0; nt < 4; nt++)
            acc[nt] = __builtin_amdgcn_wmma_f32_16x16x32_bf16(
                false, a_cur, false, b_cur[nt], (short)0, acc[nt], false, false);
        a_cur = a_nxt;
        #pragma unroll
        for (int nt = 0; nt < 4; nt++) b_cur[nt] = b_nxt[nt];
    }
    // epilogue (last K step)
    #pragma unroll
    for (int nt = 0; nt < 4; nt++)
        acc[nt] = __builtin_amdgcn_wmma_f32_16x16x32_bf16(
            false, a_cur, false, b_cur[nt], (short)0, acc[nt], false, false);

    #pragma unroll
    for (int nt = 0; nt < 4; nt++) {
        #pragma unroll
        for (int r = 0; r < 8; r++) {
            int mm = mtile * 16 + r + khalf * 8;
            int nn = n0 + nt * 16 + mloc;
            if (mm < M_total)
                *(gbf16p)(uintptr_t)&C[(size_t)mm * Nout + nn] =
                    (__bf16)((acc[nt][r] + bias[nn]) * scale);
        }
    }
}

// ---------------- Kernel: fused window cross-attention ----------------
// block = 128 threads = 4 waves; wave w handles query M-tile w of one (window, head).
__global__ __launch_bounds__(128)
void attn_kernel(const bf16* __restrict__ q, const bf16* __restrict__ kv,
                 const float* __restrict__ biasfull, bf16* __restrict__ attnout) {
    __shared__ bf16 Kt[98 * 32];        // K tile, token-major (same layout as global)
    __shared__ bf16 Vt[32][128];        // V transposed: [d][token], zero-padded tokens
    __shared__ bf16 P[4][16][128];      // softmax probs per wave, bf16, A-layout source

    const int bh = blockIdx.x;
    const int b = bh >> 4, h = bh & 15;
    const int tid = threadIdx.x, lane = tid & 31, wave = tid >> 5;
    const int mloc = lane & 15, khalf = lane >> 4;

    // Stage K tile via async global->LDS B128 copies (layout-preserving, ASYNCcnt-tracked)
    for (int idx = tid; idx < 98 * 4; idx += 128) {
        int t = idx >> 2, c = idx & 3;
        const bf16* src = kv + ((size_t)b * 98 + t) * 1024 + h * 32 + c * 8;
        unsigned dst = (unsigned)(uintptr_t)&Kt[t * 32 + c * 8];
        asm volatile("global_load_async_to_lds_b128 %0, %1, off"
                     :: "v"(dst), "v"((uint64_t)(uintptr_t)src)
                     : "memory");
    }

    // Stage V^T into LDS (tokens >= 98 zeroed so padded P columns contribute 0)
    for (int idx = tid; idx < 32 * 128; idx += 128) {
        int d = idx >> 7, t = idx & 127;
        bf16 val = (__bf16)0.0f;
        if (t < 98) val = kv[((size_t)b * 98 + t) * 1024 + 512 + h * 32 + d];
        Vt[d][t] = val;
    }
    asm volatile("s_wait_asynccnt 0x0" ::: "memory");
    __syncthreads();

    // Q fragment (A: 16x32, hd == one full K step), scale already folded in
    int m  = wave * 16 + mloc;
    int mc = min(m, 48);
    v16bf af = gbl_frag(q + ((size_t)b * 49 + mc) * 512 + h * 32 + khalf * 8);

    // S = q @ k^T over 7 key tiles (98 -> 112 padded), K fragments from LDS
    v8f s[7];
    #pragma unroll
    for (int nt = 0; nt < 7; nt++) {
        int t  = nt * 16 + mloc;
        int tc = min(t, 97);
        v16bf bf = lds_frag(&Kt[tc * 32 + khalf * 8]);
        v8f z = {};
        s[nt] = __builtin_amdgcn_wmma_f32_16x16x32_bf16(
            false, af, false, bf, (short)0, z, false, false);
    }

    // Add padded bias+mask (C layout: VGPR r -> row r+khalf*8, lane&15 -> col)
    const float* bb = biasfull + (((size_t)(b & 63) * 16 + h) * 64 + wave * 16) * 128;
    #pragma unroll
    for (int nt = 0; nt < 7; nt++)
        #pragma unroll
        for (int r = 0; r < 8; r++)
            s[nt][r] += bb[(r + khalf * 8) * 128 + nt * 16 + mloc];

    // Row softmax: reduce across 7 tiles per register, then across the 16-lane half
    float inv[8];
    #pragma unroll
    for (int r = 0; r < 8; r++) {
        float mx = -3.0e38f;
        #pragma unroll
        for (int nt = 0; nt < 7; nt++) mx = fmaxf(mx, s[nt][r]);
        #pragma unroll
        for (int off = 1; off < 16; off <<= 1) mx = fmaxf(mx, __shfl_xor(mx, off, 32));
        float sum = 0.0f;
        #pragma unroll
        for (int nt = 0; nt < 7; nt++) {
            float e = __expf(s[nt][r] - mx);
            s[nt][r] = e;
            sum += e;
        }
        #pragma unroll
        for (int off = 1; off < 16; off <<= 1) sum += __shfl_xor(sum, off, 32);
        inv[r] = 1.0f / sum;
    }

    // Write normalized P to LDS (bf16), zero the 112..127 pad columns
    #pragma unroll
    for (int nt = 0; nt < 7; nt++)
        #pragma unroll
        for (int r = 0; r < 8; r++)
            P[wave][r + khalf * 8][nt * 16 + mloc] = (__bf16)(s[nt][r] * inv[r]);
    #pragma unroll
    for (int r = 0; r < 8; r++)
        P[wave][r + khalf * 8][112 + mloc] = (__bf16)0.0f;

    // O = P @ V : 4 K-steps (128 keys) x 2 output tiles (hd=32)
    v8f o[2] = {};
    #pragma unroll
    for (int kk = 0; kk < 4; kk++) {
        v16bf pa = lds_frag(&P[wave][mloc][kk * 32 + khalf * 8]);
        #pragma unroll
        for (int dt = 0; dt < 2; dt++) {
            v16bf vb = lds_frag(&Vt[dt * 16 + mloc][kk * 32 + khalf * 8]);
            o[dt] = __builtin_amdgcn_wmma_f32_16x16x32_bf16(
                false, pa, false, vb, (short)0, o[dt], false, false);
        }
    }
    #pragma unroll
    for (int dt = 0; dt < 2; dt++)
        #pragma unroll
        for (int r = 0; r < 8; r++) {
            int mm = wave * 16 + r + khalf * 8;
            if (mm < 49)
                *(gbf16p)(uintptr_t)&attnout[((size_t)b * 49 + mm) * 512 + h * 32 + dt * 16 + mloc]
                    = (__bf16)o[dt][r];
        }
}

// ---------------- Kernel: GEMM  C_f32 = A_bf16 @ W^T + bias (output projection) ----------------
__global__ __launch_bounds__(128)
void gemm_abf16_cf32(const bf16* __restrict__ A, const bf16* __restrict__ W,
                     const float* __restrict__ bias, float* __restrict__ C,
                     int M_total, int Nout) {
    const int lane  = threadIdx.x & 31;
    const int wave  = threadIdx.x >> 5;
    const int mtile = blockIdx.x * 4 + wave;
    const int n0    = blockIdx.y * 64;
    if (mtile * 16 >= M_total) return;
    const int mloc  = lane & 15;
    const int khalf = lane >> 4;

    int mc = min(mtile * 16 + mloc, M_total - 1);
    const bf16* aptr = A + (size_t)mc * 512 + khalf * 8;
    const bf16* wptr[4];
    #pragma unroll
    for (int nt = 0; nt < 4; nt++)
        wptr[nt] = W + (size_t)(n0 + nt * 16 + mloc) * 512 + khalf * 8;

    v16bf a_cur = gbl_frag(aptr);
    v16bf b_cur[4];
    #pragma unroll
    for (int nt = 0; nt < 4; nt++) b_cur[nt] = gbl_frag(wptr[nt]);

    v8f acc[4] = {};
    #pragma unroll 3
    for (int k0 = 0; k0 < 480; k0 += 32) {
        aptr += 32;
        __builtin_prefetch(aptr + 96, 0, 1);
        v16bf a_nxt = gbl_frag(aptr);
        v16bf b_nxt[4];
        #pragma unroll
        for (int nt = 0; nt < 4; nt++) {
            wptr[nt] += 32;
            b_nxt[nt] = gbl_frag(wptr[nt]);
        }
        #pragma unroll
        for (int nt = 0; nt < 4; nt++)
            acc[nt] = __builtin_amdgcn_wmma_f32_16x16x32_bf16(
                false, a_cur, false, b_cur[nt], (short)0, acc[nt], false, false);
        a_cur = a_nxt;
        #pragma unroll
        for (int nt = 0; nt < 4; nt++) b_cur[nt] = b_nxt[nt];
    }
    #pragma unroll
    for (int nt = 0; nt < 4; nt++)
        acc[nt] = __builtin_amdgcn_wmma_f32_16x16x32_bf16(
            false, a_cur, false, b_cur[nt], (short)0, acc[nt], false, false);

    #pragma unroll
    for (int nt = 0; nt < 4; nt++)
        #pragma unroll
        for (int r = 0; r < 8; r++) {
            int mm = mtile * 16 + r + khalf * 8;
            int nn = n0 + nt * 16 + mloc;
            if (mm < M_total)
                *(gf32p)(uintptr_t)&C[(size_t)mm * Nout + nn] = acc[nt][r] + bias[nn];
        }
}

// ---------------- Host launch ----------------
extern "C" void kernel_launch(void* const* d_in, const int* in_sizes, int n_in,
                              void* d_out, int out_size, void* d_ws, size_t ws_size,
                              hipStream_t stream) {
    const float* x          = (const float*)d_in[0];   // (2048,49,512)
    const float* y          = (const float*)d_in[1];   // (2048,49,512)
    const float* wq         = (const float*)d_in[2];   // (512,512)
    const float* bq         = (const float*)d_in[3];   // (512)
    const float* wkv        = (const float*)d_in[4];   // (1024,512)
    const float* bkv        = (const float*)d_in[5];   // (1024)
    const float* wp         = (const float*)d_in[6];   // (512,512)
    const float* bp         = (const float*)d_in[7];   // (512)
    const float* bias_table = (const float*)d_in[8];   // (169,16)
    const float* mask       = (const float*)d_in[9];   // (64,49,49)
    const int*   rel_index  = (const int*)d_in[10];    // (49,49)
    float* out = (float*)d_out;

    char* ws = (char*)d_ws;
    bf16*  wq_bf    = (bf16*)(ws);                         //   512 KB
    bf16*  wkv_bf   = (bf16*)(ws + 524288);                //  1024 KB
    bf16*  wp_bf    = (bf16*)(ws + 1572864);               //   512 KB
    float* biasfull = (float*)(ws + 2097152);              //    32 MB (64,16,64,128)
    bf16*  q_bf     = (bf16*)(ws + 35651584);              //  ~98 MB (2048*49,512)
    bf16*  kv_bf    = (bf16*)(ws + 138412032);             // ~392 MB (2048*98,1024)
    bf16*  attnout  = (bf16*)(ws + 549453824);             //  ~98 MB (2048*49,512)

    const float scale = 0.17677669529663687f;  // 1/sqrt(32)

    // 1) convert weights to bf16 (tiny; L2-resident afterwards)
    cvt_f32_bf16<<<(512 * 512 + 255) / 256, 256, 0, stream>>>(wq, wq_bf, 512 * 512);
    cvt_f32_bf16<<<(1024 * 512 + 255) / 256, 256, 0, stream>>>(wkv, wkv_bf, 1024 * 512);
    cvt_f32_bf16<<<(512 * 512 + 255) / 256, 256, 0, stream>>>(wp, wp_bf, 512 * 512);

    // 2) pack padded bias+mask
    bias_pack<<<(64 * 16 * 64 * 128) / 256, 256, 0, stream>>>(bias_table, rel_index, mask, biasfull);

    // 3) q = (x @ wq^T + bq) * scale   [M = 2048*49]
    {
        int M = 2048 * 49;
        dim3 grid((M / 16 + 3) / 4, 512 / 64);
        gemm_a32_cbf16<<<grid, 128, 0, stream>>>(x, x, 49, 49, wq_bf, bq, q_bf, M, 512, scale);
    }
    // 4) kv = concat(x,y) @ wkv^T + bkv   [M = 2048*98]
    {
        int M = 2048 * 98;
        dim3 grid((M / 16 + 3) / 4, 1024 / 64);
        gemm_a32_cbf16<<<grid, 128, 0, stream>>>(x, y, 98, 49, wkv_bf, bkv, kv_bf, M, 1024, 1.0f);
    }
    // 5) fused attention per (window, head)
    attn_kernel<<<2048 * 16, 128, 0, stream>>>(q_bf, kv_bf, biasfull, attnout);

    // 6) out = attnout @ wp^T + bp
    {
        int M = 2048 * 49;
        dim3 grid((M / 16 + 3) / 4, 512 / 64);
        gemm_abf16_cf32<<<grid, 128, 0, stream>>>(attnout, wp_bf, bp, out, M, 512);
    }
}
